// LayerCorrelationCalculation_13864154431821
// MI455X (gfx1250) — compile-verified, matrix operand order checked by reading, not run on hardware
//
#include <hip/hip_runtime.h>
#include <hip/hip_bf16.h>
#include <math.h>

typedef __attribute__((ext_vector_type(16))) _Float16 v16h;
typedef __attribute__((ext_vector_type(8)))  float    v8f;

#define TILE_M 16
#define KBLK   32
#define LOG2E  1.44269504088896340736f

// ---------------------------------------------------------------------------
// Kernel 1: per-batch min/max of x (gives exact q-range since q = w0*x + b0).
// ---------------------------------------------------------------------------
__global__ __launch_bounds__(256)
void batch_minmax_kernel(const float* __restrict__ x, float* __restrict__ mm, int n) {
    __shared__ float smin[256];
    __shared__ float smax[256];
    const int b = blockIdx.x;
    const float* xb = x + (size_t)b * n;
    float lo =  3.4e38f, hi = -3.4e38f;
    for (int i = threadIdx.x; i < n; i += blockDim.x) {
        float v = xb[i];
        lo = fminf(lo, v);
        hi = fmaxf(hi, v);
    }
    smin[threadIdx.x] = lo;
    smax[threadIdx.x] = hi;
    __syncthreads();
    for (int s = 128; s > 0; s >>= 1) {
        if (threadIdx.x < s) {
            smin[threadIdx.x] = fminf(smin[threadIdx.x], smin[threadIdx.x + s]);
            smax[threadIdx.x] = fmaxf(smax[threadIdx.x], smax[threadIdx.x + s]);
        }
        __syncthreads();
    }
    if (threadIdx.x == 0) {
        mm[b * 2 + 0] = smin[0];
        mm[b * 2 + 1] = smax[0];
    }
}

// ---------------------------------------------------------------------------
// Kernel 2: fused qkv + rank-1 softmax-attention + sigmoid gate.
// One wave32 per 16-wide m-tile. Per 32-wide n-block (all-register, no LDS):
//   E [k,m] = exp2(x[n]*kw0[m] + kb0[m])          (f16, in (0,1])
//   VE[k,m] = v[n] * E[k,m]                       (f16)
//   accE = WMMA(Ones, E , accE)  ->  row0 = sum e
//   accV = WMMA(Ones, VE, accV)  ->  row0 = sum v*e
// where Ones is a constant A-tile (row 0 = 1.0, rest 0), hoisted out of loop.
// ---------------------------------------------------------------------------
__global__ __launch_bounds__(128)
void attn_gate_kernel(const float* __restrict__ x,
                      const float* __restrict__ Wq,
                      const float* __restrict__ bq,
                      const float* __restrict__ mm,
                      float* __restrict__ out, int n) {
    const int lane = threadIdx.x & 31;
    const int g    = lane >> 4;      // half-wave group: owns K = 16g..16g+15
    const int col  = lane & 15;      // owned B/C/D column, also owned A row
    const int tilesPerBatch = n / TILE_M;

    const int wid = blockIdx.x * (blockDim.x >> 5) + (threadIdx.x >> 5);
    const int b   = wid / tilesPerBatch;
    const int m0  = (wid % tilesPerBatch) * TILE_M;

    const float w0 = Wq[0], w1 = Wq[1], w2 = Wq[2];
    const float b0 = bq[0], b1 = bq[1], b2 = bq[2];

    // q-range from x-range (q affine in x, monotone per sign of w0)
    const float xlo = mm[b * 2 + 0];
    const float xhi = mm[b * 2 + 1];
    const float qmin = (w0 >= 0.f) ? fmaf(w0, xlo, b0) : fmaf(w0, xhi, b0);
    const float qmax = (w0 >= 0.f) ? fmaf(w0, xhi, b0) : fmaf(w0, xlo, b0);

    const float* xb = x + (size_t)b * n;

    const float xm   = xb[m0 + col];                    // x at this lane's m
    const float kcol = fmaf(w1, xm, b1);                // k[m]
    const float ccol = fmaxf(qmax * kcol, qmin * kcol); // max_n q[n]*k[m]
    // fold q-affine + max-shift + log2e into one FMA per element:
    //   t = log2e*(q[n]*k[m] - c[m]) = x[n]*kw0 + kb0
    const float kw0 = w0 * kcol * LOG2E;
    const float kb0 = (b0 * kcol - ccol) * LOG2E;

    // constant A-tile: row 0 = all ones (lane col==0 in both half-waves)
    const _Float16 arow = (col == 0) ? (_Float16)1.0f : (_Float16)0.0f;
    v16h ones;
    #pragma unroll
    for (int h = 0; h < 16; ++h) ones[h] = arow;

    v8f accE = {};
    v8f accV = {};

    // this lane's contiguous 16-float K-run within each 32-block
    const float* pK = xb + 16 * g;

    for (int n0 = 0; n0 < n; n0 += KBLK) {
        const float4 c0 = *(const float4*)(pK + n0 + 0);
        const float4 c1 = *(const float4*)(pK + n0 + 4);
        const float4 c2 = *(const float4*)(pK + n0 + 8);
        const float4 c3 = *(const float4*)(pK + n0 + 12);
        float xs[16] = { c0.x, c0.y, c0.z, c0.w,  c1.x, c1.y, c1.z, c1.w,
                         c2.x, c2.y, c2.z, c2.w,  c3.x, c3.y, c3.z, c3.w };

        v16h emat, vmat;
        #pragma unroll
        for (int h = 0; h < 16; ++h) {
            const float e = __builtin_amdgcn_exp2f(fmaf(xs[h], kw0, kb0));
            const float v = fmaf(w2, xs[h], b2);
            emat[h] = (_Float16)e;
            vmat[h] = (_Float16)(v * e);
        }

        accE = __builtin_amdgcn_wmma_f32_16x16x32_f16(
                   false, ones, false, emat, (short)0, accE, false, false);
        accV = __builtin_amdgcn_wmma_f32_16x16x32_f16(
                   false, ones, false, vmat, (short)0, accV, false, false);
    }

    // D layout: lanes 0-15, VGPR0 = row M=0 -> accE[0]=sum e, accV[0]=sum v*e
    if (g == 0) {
        const float o  = accV[0] * __builtin_amdgcn_rcpf(accE[0]);
        const float sg = __builtin_amdgcn_rcpf(
                             1.0f + __builtin_amdgcn_exp2f(-o * LOG2E));
        out[(size_t)b * n + m0 + col] = xm * sg;
    }
}

// ---------------------------------------------------------------------------
extern "C" void kernel_launch(void* const* d_in, const int* in_sizes, int n_in,
                              void* d_out, int out_size, void* d_ws, size_t ws_size,
                              hipStream_t stream) {
    const float* x  = (const float*)d_in[0];   // [B,1,96,96] f32
    const float* Wq = (const float*)d_in[1];   // [3,1] f32
    const float* bq = (const float*)d_in[2];   // [3]   f32
    float* out = (float*)d_out;                // [B,1,96,96] f32
    float* mm  = (float*)d_ws;                 // per-batch {min,max}

    const int n = 96 * 96;                     // N = 9216 (multiple of 32)
    const int B = in_sizes[0] / n;             // 2

    batch_minmax_kernel<<<B, 256, 0, stream>>>(x, mm, n);

    const int totalWaves = B * (n / TILE_M);   // 1152 wave32's
    const int blocks     = totalWaves / 4;     // 4 waves (128 thr) per block
    attn_gate_kernel<<<blocks, 128, 0, stream>>>(x, Wq, bq, mm, out, n);
}
